// TextCnnClassifier_56014963474478
// MI455X (gfx1250) — matrix operand order repaired and checked by reading, not compiled, not run on hardware
//
#include <hip/hip_runtime.h>

typedef __attribute__((ext_vector_type(4)))  unsigned int v4u;
typedef __attribute__((ext_vector_type(2)))  unsigned int v2u;
typedef __attribute__((ext_vector_type(16))) __bf16      v16bf;
typedef __attribute__((ext_vector_type(8)))  float       v8f;

union ABFrag {
    v4u   u4[2];
    v2u   u2[4];
    v16bf bf;
};

__device__ __forceinline__ unsigned short f2bf(float f) {
    unsigned int u = __float_as_uint(f);
    unsigned int r = u + 0x7FFFu + ((u >> 16) & 1u);   // round-to-nearest-even
    return (unsigned short)(r >> 16);
}

// ---- pre-convert x [16*1024*300] fp32 -> bf16 into workspace, zero pad tail ----
__global__ void cvt_x_bf16(const float* __restrict__ x,
                           unsigned short* __restrict__ xbf,
                           int nElem, int nPad) {
    int base = (blockIdx.x * blockDim.x + threadIdx.x) * 8;
    if (base < nElem) {
        union { v4u u4; unsigned short s[8]; } o;
        const float* p = x + base;
#pragma unroll
        for (int i = 0; i < 8; ++i) o.s[i] = f2bf(p[i]);
        *(v4u*)(xbf + base) = o.u4;
    } else if (base < nPad) {
        v4u z = {};
        *(v4u*)(xbf + base) = z;
    }
}

// Per conv kernel-size KS: for one batch b and BM output channels,
//   y[m,t] = sum_kk A[m,kk] * xwin[kk + j*300],   A[m,dk*300+c] = W[m,c,dk]
//   out[b,o,off+n] = relu(bias[m] + max_t y[m,t]),  m = n*100 + o.
// Each wave: 16 M rows x 32 t columns per chunk (2 accumulators, A reused).
// x window double-buffered in LDS; staged via async-to-LDS when x is
// pre-converted to bf16 in the workspace (PRECVT), else fp32 load+convert.
template<int KS, int KPAD, int BM, bool PRECVT>
__global__ __launch_bounds__((BM / 16) * 32)
void convmax_kernel(const float* __restrict__ x,             // [16,1024,300] fp32
                    const unsigned short* __restrict__ xbf,  // [16*307200+pad] bf16
                    const float* __restrict__ W,             // [6000,300,KS]
                    const float* __restrict__ bias,          // [6000]
                    float* __restrict__ out,                 // [16,100,180]
                    int branchOff)
{
    constexpr int K      = 300 * KS;
    constexpr int T      = 1025 - KS;
    constexpr int NT     = (BM / 16) * 32;
    constexpr int WINR   = KPAD + 31 * 300;          // window for 32 positions
    constexpr int WIN    = (WINR + 7) & ~7;          // 16B-unit padded
    constexpr int NCHUNK = (T + 31) / 32;            // = 32
    constexpr int NKC    = KPAD / 32;

    __shared__ __align__(16) unsigned short sA[BM * KPAD];
    __shared__ __align__(16) unsigned short sX[2][WIN];

    const int tid   = threadIdx.x;
    const int lane  = tid & 31;
    const int wave  = tid >> 5;
    const int jlane = lane & 15;
    const int hi    = (lane >> 4);

    const int b     = blockIdx.y;
    const int mBase = blockIdx.x * BM;
    const float* xB = x + b * 307200;

    // ---- stage A: transpose (c,dk)->(dk*300+c), fp32->bf16, zero-pad K ----
    for (int idx = tid; idx < BM * KPAD; idx += NT) {
        int ml = idx / KPAD;
        int kk = idx - ml * KPAD;
        int m  = mBase + ml;
        float v = 0.0f;
        if (kk < K && m < 6000) {
            int dk = kk / 300;
            int c  = kk - dk * 300;
            v = W[m * K + c * KS + dk];
        }
        sA[idx] = f2bf(v);
    }

    // stage the x window for chunk tc into buffer buf
    auto stage = [&](int tc, int buf) {
        const int t0 = tc * 32;
        if constexpr (PRECVT) {
            const unsigned short* gsrc = xbf + (size_t)b * 307200;  // uniform SGPR base
            unsigned ldsBase = (unsigned)(unsigned long long)&sX[buf][0];
            unsigned gByte   = (unsigned)(t0 * 600);
            for (int u = tid; u < WIN / 8; u += NT) {
                unsigned lds  = ldsBase + (unsigned)u * 16u;
                unsigned voff = gByte + (unsigned)u * 16u;
                asm volatile("global_load_async_to_lds_b128 %0, %1, %2 offset:0"
                             :: "v"(lds), "v"(voff), "s"(gsrc) : "memory");
            }
        } else {
            const int gbase = t0 * 300;
            for (int i = tid; i < WIN; i += NT) {
                int g = gbase + i;
                float v = (g < 307200) ? xB[g] : 0.0f;
                sX[buf][i] = f2bf(v);
            }
        }
    };

    const unsigned short* aRow = sA + (wave * 16 + jlane) * KPAD;

    float rm[8];
#pragma unroll
    for (int r = 0; r < 8; ++r) rm[r] = -3.4e38f;

    stage(0, 0);                                   // prologue: window 0 in flight

    for (int tc = 0; tc < NCHUNK; ++tc) {
        if constexpr (PRECVT)
            asm volatile("s_wait_asynccnt 0" ::: "memory");
        __syncthreads();                           // buf[tc&1] ready; buf[!(tc&1)] reads done
        if (tc + 1 < NCHUNK) stage(tc + 1, (tc + 1) & 1);

        const unsigned short* sXc = sX[tc & 1];
        v8f acc0 = {}, acc1 = {};
#pragma unroll 4
        for (int kc = 0; kc < NKC; ++kc) {
            ABFrag a;                               // A: M=jlane, K={0..7,16..23}(+8 if hi)
            const int aoff = kc * 32 + hi * 8;
            a.u4[0] = *(const v4u*)(aRow + aoff);
            a.u4[1] = *(const v4u*)(aRow + aoff + 16);

            const int boff = kc * 32 + hi * 8 + jlane * 300;
            ABFrag b0, b1;                          // B[kk][j] = sX[kk + j*300]
            const v2u* p0 = (const v2u*)(sXc + boff);
            b0.u2[0] = p0[0]; b0.u2[1] = p0[1]; b0.u2[2] = p0[4]; b0.u2[3] = p0[5];
            const v2u* p1 = (const v2u*)(sXc + boff + 4800);  // +16 positions
            b1.u2[0] = p1[0]; b1.u2[1] = p1[1]; b1.u2[2] = p1[4]; b1.u2[3] = p1[5];

            acc0 = __builtin_amdgcn_wmma_f32_16x16x32_bf16(
                       false, a.bf, false, b0.bf, (short)0, acc0, false, false);
            acc1 = __builtin_amdgcn_wmma_f32_16x16x32_bf16(
                       false, a.bf, false, b1.bf, (short)0, acc1, false, false);
        }

        const int t0 = tc * 32;
        if (t0 + jlane < T) {
#pragma unroll
            for (int r = 0; r < 8; ++r) rm[r] = fmaxf(rm[r], acc0[r]);
        }
        if (t0 + 16 + jlane < T) {
#pragma unroll
            for (int r = 0; r < 8; ++r) rm[r] = fmaxf(rm[r], acc1[r]);
        }
    }

    // ---- max over the 16 t-columns in each 16-lane half ----
#pragma unroll
    for (int mask = 8; mask >= 1; mask >>= 1) {
#pragma unroll
        for (int r = 0; r < 8; ++r)
            rm[r] = fmaxf(rm[r], __shfl_xor(rm[r], mask, 32));
    }

    if (jlane == 0) {                  // lanes 0/16 hold M=r / M=8+r
        const int mRow = mBase + wave * 16 + hi * 8;
#pragma unroll
        for (int r = 0; r < 8; ++r) {
            int m = mRow + r;
            if (m < 6000) {
                float v = fmaxf(rm[r] + bias[m], 0.0f);
                int n = m / 100;
                int o = m - n * 100;
                out[b * (100 * 180) + o * 180 + branchOff + n] = v;
            }
        }
    }
}

extern "C" void kernel_launch(void* const* d_in, const int* in_sizes, int n_in,
                              void* d_out, int out_size, void* d_ws, size_t ws_size,
                              hipStream_t stream) {
    const float* x  = (const float*)d_in[0];
    const float* W2 = (const float*)d_in[1];
    const float* b2 = (const float*)d_in[2];
    const float* W3 = (const float*)d_in[3];
    const float* b3 = (const float*)d_in[4];
    const float* W4 = (const float*)d_in[5];
    const float* b4 = (const float*)d_in[6];
    float* out = (float*)d_out;

    constexpr int N_X   = 16 * 307200;
    constexpr int N_PAD = N_X + 32768;
    unsigned short* xbf = (unsigned short*)d_ws;
    const bool pre = ws_size >= (size_t)N_PAD * sizeof(unsigned short);

    if (pre) {
        int units = N_PAD / 8;
        cvt_x_bf16<<<(units + 255) / 256, 256, 0, stream>>>(x, xbf, N_X, N_PAD);
        convmax_kernel<2,  608, 96, true><<<dim3( 63, 16), 192, 0, stream>>>(x, xbf, W2, b2, out,   0);
        convmax_kernel<3,  928, 64, true><<<dim3( 94, 16), 128, 0, stream>>>(x, xbf, W3, b3, out,  60);
        convmax_kernel<4, 1216, 48, true><<<dim3(125, 16),  96, 0, stream>>>(x, xbf, W4, b4, out, 120);
    } else {
        convmax_kernel<2,  608, 96, false><<<dim3( 63, 16), 192, 0, stream>>>(x, xbf, W2, b2, out,   0);
        convmax_kernel<3,  928, 64, false><<<dim3( 94, 16), 128, 0, stream>>>(x, xbf, W3, b3, out,  60);
        convmax_kernel<4, 1216, 48, false><<<dim3(125, 16),  96, 0, stream>>>(x, xbf, W4, b4, out, 120);
    }
}
